// VanillaAttention_49143015801134
// MI455X (gfx1250) — matrix operand order repaired and checked by reading, not compiled, not run on hardware
//
#include <hip/hip_runtime.h>

typedef __bf16 bf16_t;
typedef __attribute__((ext_vector_type(8)))  __bf16 v8bf;
typedef __attribute__((ext_vector_type(16))) __bf16 v16bf;
typedef __attribute__((ext_vector_type(8)))  float  v8f;

#define BATCH 16
#define T_SEQ 784
#define TPAD  800
#define EMB   512
#define NH    8
#define HD    64
#define BT    (BATCH * T_SEQ)        // 12544

// ---- workspace layout (units: bf16 elements) ----
static constexpr size_t OFF_XB  = 0;                          // 12544*512
static constexpr size_t SZ_XB   = (size_t)BT * EMB;           // 6422528
static constexpr size_t SZ_W    = (size_t)EMB * EMB;          // 262144
static constexpr size_t OFF_WTQ = OFF_XB  + SZ_XB;
static constexpr size_t OFF_WTK = OFF_WTQ + SZ_W;
static constexpr size_t OFF_WTV = OFF_WTK + SZ_W;
static constexpr size_t OFF_WTO = OFF_WTV + SZ_W;
static constexpr size_t OFF_QB  = OFF_WTO + SZ_W;             // [16][8][784][64]
static constexpr size_t SZ_QB   = (size_t)BATCH * NH * T_SEQ * HD;
static constexpr size_t OFF_KB  = OFF_QB + SZ_QB;             // [16][8][800][64]
static constexpr size_t SZ_KB   = (size_t)BATCH * NH * TPAD * HD;
static constexpr size_t OFF_VT  = OFF_KB + SZ_KB;             // [16][8][64][800]
static constexpr size_t OFF_CTX = OFF_VT + SZ_KB;             // [12544][512]

// ---------- helpers ----------
static __device__ __forceinline__ v8f vzero() {
  v8f v;
#pragma unroll
  for (int i = 0; i < 8; ++i) v[i] = 0.0f;
  return v;
}

// A-operand (16x32 bf16): two contiguous 8-half (16B) chunks per lane.
static __device__ __forceinline__ v16bf load_a16(const bf16_t* p0, const bf16_t* p1) {
  union { v16bf v; v8bf h[2]; } u;
  u.h[0] = *reinterpret_cast<const v8bf*>(p0);
  u.h[1] = *reinterpret_cast<const v8bf*>(p1);
  return u.v;
}

// B-operand (32x16 bf16): 16 contiguous halves (32B-aligned) per lane.
static __device__ __forceinline__ v16bf load_b16(const bf16_t* p) {
  return *reinterpret_cast<const v16bf*>(p);
}

static __device__ __forceinline__ v8f wmma_bf16(v16bf a, v16bf b, v8f c) {
  return __builtin_amdgcn_wmma_f32_16x16x32_bf16(false, a, false, b, (short)0, c,
                                                 false, false);
}

// ---------- kernel 1: fp32 -> bf16 conversions (+ weight transposes) ----------
__global__ __launch_bounds__(256) void cvt_kernel(
    const float* __restrict__ x,
    const float* __restrict__ Wq, const float* __restrict__ Wk,
    const float* __restrict__ Wv, const float* __restrict__ Wo,
    bf16_t* __restrict__ xb,
    bf16_t* __restrict__ Wtq, bf16_t* __restrict__ Wtk,
    bf16_t* __restrict__ Wtv, bf16_t* __restrict__ Wto) {
  const size_t total = SZ_XB + 4 * SZ_W;
  for (size_t idx = (size_t)blockIdx.x * blockDim.x + threadIdx.x; idx < total;
       idx += (size_t)gridDim.x * blockDim.x) {
    if (idx < SZ_XB) {
      xb[idx] = (bf16_t)x[idx];
    } else {
      size_t r   = idx - SZ_XB;
      int    w   = (int)(r / SZ_W);
      int    e   = (int)(r % SZ_W);
      int    n   = e >> 9;          // output row (transposed)
      int    k   = e & 511;         // output col
      const float* W = (w == 0) ? Wq : (w == 1) ? Wk : (w == 2) ? Wv : Wo;
      bf16_t* Wt     = (w == 0) ? Wtq : (w == 1) ? Wtk : (w == 2) ? Wtv : Wto;
      Wt[(size_t)n * EMB + k] = (bf16_t)W[(size_t)k * EMB + n];  // Wt[n][k] = W[k][n]
    }
  }
}

// ---------- kernel 2: zero the t=784..799 pad of K and V^T ----------
__global__ __launch_bounds__(256) void pad_kernel(bf16_t* __restrict__ Kb,
                                                  bf16_t* __restrict__ Vt) {
  int i = blockIdx.x * blockDim.x + threadIdx.x;  // 128*16*64 = 131072 jobs
  if (i < BATCH * NH * (TPAD - T_SEQ) * HD) {
    int d    = i & 63;
    int rest = i >> 6;
    int t    = T_SEQ + (rest & 15);
    int bh   = rest >> 4;
    Kb[((size_t)bh * TPAD + t) * HD + d]      = (bf16_t)0.0f;
    Vt[((size_t)bh * HD + d) * TPAD + t]      = (bf16_t)0.0f;
  }
}

// ---------- kernel 3: QKV projection GEMMs (bf16 WMMA, f32 acc) ----------
// one wave = 16x64 output tile; jobs = 3 * 784 * 8 = 18816; 4 waves/block
__global__ __launch_bounds__(128) void qkv_kernel(
    const bf16_t* __restrict__ xb,
    const bf16_t* __restrict__ Wtq, const bf16_t* __restrict__ Wtk,
    const bf16_t* __restrict__ Wtv,
    const float* __restrict__ bq, const float* __restrict__ bk,
    const float* __restrict__ bv,
    bf16_t* __restrict__ Qb, bf16_t* __restrict__ Kb, bf16_t* __restrict__ Vt) {
  const int wave = threadIdx.x >> 5;
  const int lane = threadIdx.x & 31;
  const int job  = blockIdx.x * 4 + wave;       // exactly 18816
  const int mat  = job / (49 * 16 * 8);         // 0:Q 1:K 2:V
  const int rem  = job % (49 * 16 * 8);
  const int mt   = rem >> 3;                    // 0..783 (16-row tile of BT)
  const int n64  = rem & 7;                     // 0..7   (64-col slab)
  const bf16_t* Wt   = (mat == 0) ? Wtq : (mat == 1) ? Wtk : Wtv;
  const float*  bias = (mat == 0) ? bq  : (mat == 1) ? bk  : bv;

  const int lhalf = lane >> 4, lmod = lane & 15;
  const int m0 = mt * 16;
  const bf16_t* arow = xb + (size_t)(m0 + lmod) * EMB;

  v8f acc[4];
#pragma unroll
  for (int nt = 0; nt < 4; ++nt) acc[nt] = vzero();

  for (int k0 = 0; k0 < EMB; k0 += 32) {
    v16bf a = load_a16(arow + k0 + lhalf * 16, arow + k0 + lhalf * 16 + 8);
#pragma unroll
    for (int nt = 0; nt < 4; ++nt) {
      const bf16_t* brow =
          Wt + (size_t)(n64 * 64 + nt * 16 + lmod) * EMB + k0 + lhalf * 16;
      acc[nt] = wmma_bf16(a, load_b16(brow), acc[nt]);
    }
  }

  const int bidx  = m0 / T_SEQ;         // all 16 rows in one batch (784 % 16 == 0)
  const int tbase = m0 - bidx * T_SEQ;
#pragma unroll
  for (int nt = 0; nt < 4; ++nt) {
    const int col = n64 * 64 + nt * 16 + lmod;
    const float bb = bias[col];
    const int h = col >> 6, d = col & 63;
#pragma unroll
    for (int r = 0; r < 8; ++r) {
      const int t = tbase + r + 8 * lhalf;
      const float v = acc[nt][r] + bb;
      if (mat == 0) {      // Q, pre-scaled by 1/sqrt(64) (exact in bf16)
        Qb[(((size_t)bidx * NH + h) * T_SEQ + t) * HD + d] = (bf16_t)(v * 0.125f);
      } else if (mat == 1) {
        Kb[(((size_t)bidx * NH + h) * TPAD + t) * HD + d] = (bf16_t)v;
      } else {             // V stored transposed [b][h][d][t]
        Vt[(((size_t)bidx * NH + h) * HD + d) * TPAD + t] = (bf16_t)v;
      }
    }
  }
}

// ---------- kernel 4: flash attention (causal), one wave per 16-query tile ----------
// jobs = 16*8*49 = 6272; 4 waves/block -> 1568 blocks
__global__ __launch_bounds__(128) void attn_kernel(
    const bf16_t* __restrict__ Qb, const bf16_t* __restrict__ Kb,
    const bf16_t* __restrict__ Vt, bf16_t* __restrict__ ctxb) {
  __shared__ __align__(32) bf16_t plds[4][16 * 32];   // P tile per wave

  const int wave = threadIdx.x >> 5;
  const int lane = threadIdx.x & 31;
  const int job  = blockIdx.x * 4 + wave;   // exactly 6272
  const int bh   = job / 49;                // (b*8 + h)
  const int qt   = job % 49;
  const int q0   = qt * 16;
  const int lhalf = lane >> 4, lmod = lane & 15;

  const bf16_t* qbase = Qb + (size_t)bh * T_SEQ * HD;
  const bf16_t* kbase = Kb + (size_t)bh * TPAD * HD;
  const bf16_t* vbase = Vt + (size_t)bh * HD * TPAD;

  // Q tile as A-operand, K(d)=0..31 and 32..63
  const bf16_t* qrp = qbase + (size_t)(q0 + lmod) * HD;
  const v16bf aq0 = load_a16(qrp + lhalf * 16,      qrp + lhalf * 16 + 8);
  const v16bf aq1 = load_a16(qrp + 32 + lhalf * 16, qrp + 40 + lhalf * 16);

  v8f acc[4];
  float m[8], l[8];
#pragma unroll
  for (int nt = 0; nt < 4; ++nt) acc[nt] = vzero();
#pragma unroll
  for (int r = 0; r < 8; ++r) { m[r] = -1e30f; l[r] = 0.0f; }

  const int qmax = q0 + 15;
  for (int c0 = 0; c0 <= qmax; c0 += 32) {
    // ---- S = (Q*0.125) @ K^T for 32 keys (two 16-key sub-tiles) ----
    v8f S[2];
#pragma unroll
    for (int s = 0; s < 2; ++s) {
      const bf16_t* krp = kbase + (size_t)(c0 + s * 16 + lmod) * HD;
      v8f sc = vzero();
      sc = wmma_bf16(aq0, load_b16(krp + lhalf * 16),      sc);
      sc = wmma_bf16(aq1, load_b16(krp + 32 + lhalf * 16), sc);
      S[s] = sc;
    }

    // ---- causal mask + online softmax (fp32), rows live in 16-lane halves ----
    const int kk0 = c0 + lmod;
    const int kk1 = kk0 + 16;
    float rowscale[8];
#pragma unroll
    for (int r = 0; r < 8; ++r) {
      const int q = q0 + r + 8 * lhalf;
      float s0 = (kk0 <= q) ? S[0][r] : -1e30f;
      float s1 = (kk1 <= q) ? S[1][r] : -1e30f;
      float mx = fmaxf(s0, s1);
#pragma unroll
      for (int d = 1; d < 16; d <<= 1) mx = fmaxf(mx, __shfl_xor(mx, d, 32));
      const float mnew = fmaxf(m[r], mx);
      const float sc   = __expf(m[r] - mnew);
      m[r] = mnew;
      const float p0 = __expf(s0 - mnew);
      const float p1 = __expf(s1 - mnew);
      float rs = p0 + p1;
#pragma unroll
      for (int d = 1; d < 16; d <<= 1) rs += __shfl_xor(rs, d, 32);
      l[r] = l[r] * sc + rs;
      rowscale[r] = sc;
      // stage P (bf16) row-major into LDS: row = r+8*lhalf, cols lmod / lmod+16
      const int row = r + 8 * lhalf;
      plds[wave][row * 32 + lmod]      = (bf16_t)p0;
      plds[wave][row * 32 + 16 + lmod] = (bf16_t)p1;
    }
#pragma unroll
    for (int nt = 0; nt < 4; ++nt)
#pragma unroll
      for (int r = 0; r < 8; ++r) acc[nt][r] *= rowscale[r];

    // wave-internal LDS fence: stores -> loads (cross-lane, same wave)
    asm volatile("s_wait_dscnt 0x0" ::: "memory");

    // P as A-operand from LDS
    const bf16_t* prp = &plds[wave][lmod * 32];
    const v16bf ap = load_a16(prp + lhalf * 16, prp + lhalf * 16 + 8);

    // ---- ctx += P @ V  (B-operand contiguous thanks to V^T layout) ----
#pragma unroll
    for (int nt = 0; nt < 4; ++nt) {
      const bf16_t* vrp =
          vbase + (size_t)(nt * 16 + lmod) * TPAD + c0 + lhalf * 16;
      acc[nt] = wmma_bf16(ap, load_b16(vrp), acc[nt]);
    }
  }

  // ---- normalize and write ctx (bf16, [b][t][h*64+d]) ----
  float inv[8];
#pragma unroll
  for (int r = 0; r < 8; ++r) inv[r] = 1.0f / l[r];
  const int b = bh >> 3, h = bh & 7;
#pragma unroll
  for (int nt = 0; nt < 4; ++nt)
#pragma unroll
    for (int r = 0; r < 8; ++r) {
      const int t = q0 + r + 8 * lhalf;
      ctxb[((size_t)(b * T_SEQ + t)) * EMB + h * HD + nt * 16 + lmod] =
          (bf16_t)(acc[nt][r] * inv[r]);
    }
}

// ---------- kernel 5: output projection, fp32 result ----------
// jobs = 784 * 8 = 6272; 4 waves/block
__global__ __launch_bounds__(128) void oproj_kernel(
    const bf16_t* __restrict__ ctxb, const bf16_t* __restrict__ Wto,
    const float* __restrict__ bo, float* __restrict__ out) {
  const int wave = threadIdx.x >> 5;
  const int lane = threadIdx.x & 31;
  const int job  = blockIdx.x * 4 + wave;   // exactly 6272
  const int mt = job >> 3, n64 = job & 7;
  const int lhalf = lane >> 4, lmod = lane & 15;
  const int m0 = mt * 16;
  const bf16_t* arow = ctxb + (size_t)(m0 + lmod) * EMB;

  v8f acc[4];
#pragma unroll
  for (int nt = 0; nt < 4; ++nt) acc[nt] = vzero();

  for (int k0 = 0; k0 < EMB; k0 += 32) {
    v16bf a = load_a16(arow + k0 + lhalf * 16, arow + k0 + lhalf * 16 + 8);
#pragma unroll
    for (int nt = 0; nt < 4; ++nt) {
      const bf16_t* brow =
          Wto + (size_t)(n64 * 64 + nt * 16 + lmod) * EMB + k0 + lhalf * 16;
      acc[nt] = wmma_bf16(a, load_b16(brow), acc[nt]);
    }
  }

#pragma unroll
  for (int nt = 0; nt < 4; ++nt) {
    const int col = n64 * 64 + nt * 16 + lmod;
    const float bb = bo[col];
#pragma unroll
    for (int r = 0; r < 8; ++r) {
      const int gr = m0 + r + 8 * lhalf;
      out[(size_t)gr * EMB + col] = acc[nt][r] + bb;
    }
  }
}

// ---------------- launch ----------------
extern "C" void kernel_launch(void* const* d_in, const int* in_sizes, int n_in,
                              void* d_out, int out_size, void* d_ws, size_t ws_size,
                              hipStream_t stream) {
  (void)in_sizes; (void)n_in; (void)out_size; (void)ws_size;
  const float* x  = (const float*)d_in[0];
  const float* Wq = (const float*)d_in[1];
  const float* bq = (const float*)d_in[2];
  const float* Wk = (const float*)d_in[3];
  const float* bk = (const float*)d_in[4];
  const float* Wv = (const float*)d_in[5];
  const float* bv = (const float*)d_in[6];
  const float* Wo = (const float*)d_in[7];
  const float* bo = (const float*)d_in[8];
  float* out = (float*)d_out;

  bf16_t* ws  = (bf16_t*)d_ws;
  bf16_t* xb  = ws + OFF_XB;
  bf16_t* Wtq = ws + OFF_WTQ;
  bf16_t* Wtk = ws + OFF_WTK;
  bf16_t* Wtv = ws + OFF_WTV;
  bf16_t* Wto = ws + OFF_WTO;
  bf16_t* Qb  = ws + OFF_QB;
  bf16_t* Kb  = ws + OFF_KB;
  bf16_t* Vt  = ws + OFF_VT;
  bf16_t* ctx = ws + OFF_CTX;

  cvt_kernel<<<4096, 256, 0, stream>>>(x, Wq, Wk, Wv, Wo, xb, Wtq, Wtk, Wtv, Wto);
  pad_kernel<<<512, 256, 0, stream>>>(Kb, Vt);
  qkv_kernel<<<4704, 128, 0, stream>>>(xb, Wtq, Wtk, Wtv, bq, bk, bv, Qb, Kb, Vt);
  attn_kernel<<<1568, 128, 0, stream>>>(Qb, Kb, Vt, ctx);
  oproj_kernel<<<1568, 128, 0, stream>>>(ctx, Wto, bo, out);
}
